// JCAF_66709432042121
// MI455X (gfx1250) — compile-verified, matrix operand order checked by reading, not run on hardware
//
// JCAF fusion for MI455X (gfx1250, wave32, WMMA), round 3.
// Bandwidth-bound (~0.5 GB traffic -> ~20us @ 23.3 TB/s). All contractions on
// v_wmma_f32_16x16x32_bf16. Operands stored row-major along K: fragments are
// two aligned global_load_b128 per lane. Heavy GEMMs use 2x2 register tiling
// (4 wmma per 4 fragment loads) + software-pipelined k-loop (loads for k+32
// issued before the wmmas for k, so waits overlap with compute).

#include <hip/hip_runtime.h>
#include <hip/hip_bf16.h>
#include <stdint.h>

typedef __bf16 bf16;
typedef __attribute__((ext_vector_type(16))) __bf16 v16bf;
typedef __attribute__((ext_vector_type(8)))  __bf16 v8bf;
typedef __attribute__((ext_vector_type(4)))  __bf16 v4bf;
typedef __attribute__((ext_vector_type(8)))  float  v8f;

#define B_  64
#define S_  1024
#define F_  128
#define F2_ 256
#define K_  64

// ---------- scalar bf16 helper (RNE) ----------
__device__ __forceinline__ bf16 f2bf(float x) {
  unsigned u = __builtin_bit_cast(unsigned, x);
  unsigned r = (u + 0x7FFFu + ((u >> 16) & 1u)) >> 16;
  unsigned short s = (unsigned short)r;
  return __builtin_bit_cast(bf16, s);
}

// ---------- WMMA fragment loader: operand stored [rows, K] row-major ----------
// Per lane: row = row0 + (lane&15); two 16B loads at k0 + half*8 (+16).
// K(e) = (e&7) | (half<<3) | ((e&8)<<1) => e0..7 / e8..15 K-contiguous.
__device__ __forceinline__ v16bf load_frag(const bf16* __restrict__ P, int ld,
                                           int row0, int k0) {
  int lane = threadIdx.x & 31;
  int r = lane & 15, hf = lane >> 4;
  const bf16* p = P + (size_t)(row0 + r) * ld + k0 + hf * 8;
  v8bf lo = *(const v8bf*)p;
  v8bf hi = *(const v8bf*)(p + 16);
  v16bf out;
#pragma unroll
  for (int e = 0; e < 8; ++e) { out[e] = lo[e]; out[e + 8] = hi[e]; }
  return out;
}

__device__ __forceinline__ v8f wmma_bf16(v16bf a, v16bf b, v8f c) {
  return __builtin_amdgcn_wmma_f32_16x16x32_bf16(false, a, false, b,
                                                 (short)0, c, false, false);
}

// ---------- pipelined 1x1 GEMM tile: acc += A[row0:16, :] * Bt[col0:16, :]^T --
__device__ __forceinline__ v8f gemm_1x1(const bf16* __restrict__ A, int lda,
                                        int row0, const bf16* __restrict__ Bt,
                                        int ldb, int col0, int klen, v8f acc) {
  v16bf a0 = load_frag(A, lda, row0, 0);
  v16bf b0 = load_frag(Bt, ldb, col0, 0);
  for (int k0 = 0; k0 < klen; k0 += 32) {
    int kn = (k0 + 32 < klen) ? (k0 + 32) : k0;  // prefetch next (dup on last)
    v16bf a1 = load_frag(A, lda, row0, kn);
    v16bf b1 = load_frag(Bt, ldb, col0, kn);
    acc = wmma_bf16(a0, b0, acc);
    a0 = a1; b0 = b1;
  }
  return acc;
}

// ---------- pipelined 2x2 GEMM tile: 32x32 output, 4 wmma per 4 frag loads ---
__device__ __forceinline__ void gemm_2x2(const bf16* __restrict__ A, int lda,
                                         int row0, const bf16* __restrict__ Bt,
                                         int ldb, int col0, int klen,
                                         v8f acc[4]) {
  v16bf a0 = load_frag(A, lda, row0, 0);
  v16bf a1 = load_frag(A, lda, row0 + 16, 0);
  v16bf b0 = load_frag(Bt, ldb, col0, 0);
  v16bf b1 = load_frag(Bt, ldb, col0 + 16, 0);
  for (int k0 = 0; k0 < klen; k0 += 32) {
    int kn = (k0 + 32 < klen) ? (k0 + 32) : k0;
    v16bf a0n = load_frag(A, lda, row0, kn);
    v16bf a1n = load_frag(A, lda, row0 + 16, kn);
    v16bf b0n = load_frag(Bt, ldb, col0, kn);
    v16bf b1n = load_frag(Bt, ldb, col0 + 16, kn);
    acc[0] = wmma_bf16(a0, b0, acc[0]);
    acc[1] = wmma_bf16(a0, b1, acc[1]);
    acc[2] = wmma_bf16(a1, b0, acc[2]);
    acc[3] = wmma_bf16(a1, b1, acc[3]);
    a0 = a0n; a1 = a1n; b0 = b0n; b1 = b1n;
  }
}

__device__ __forceinline__ void store_tile_bf16(bf16* __restrict__ D, int ld,
                                                int row0, int col0, v8f c) {
  int lane = threadIdx.x & 31, n = lane & 15, hf = lane >> 4;
#pragma unroll
  for (int r = 0; r < 8; ++r)
    D[(size_t)(row0 + r + hf * 8) * ld + col0 + n] = f2bf(c[r]);
}

// ---------- plain f32 -> bf16 convert (vectorized x4) ----------
__global__ __launch_bounds__(256) void k_cvt_plain(const float* __restrict__ src,
                                                   bf16* __restrict__ dst,
                                                   long n4) {
  const float4* s4 = (const float4*)src;
  v4bf* d4 = (v4bf*)dst;
  for (long i = blockIdx.x * (long)blockDim.x + threadIdx.x; i < n4;
       i += (long)gridDim.x * blockDim.x) {
    float4 v = s4[i];
    v4bf o;
    o[0] = f2bf(v.x); o[1] = f2bf(v.y); o[2] = f2bf(v.z); o[3] = f2bf(v.w);
    d4[i] = o;
  }
}

// ---------- batched 32x32 LDS-tiled transpose: f32 [nb,R,C] -> bf16 [nb,C,R] --
__global__ __launch_bounds__(256) void k_tr_cvt(const float* __restrict__ src,
                                                bf16* __restrict__ dst,
                                                int R, int C) {
  __shared__ float tile[32][33];
  int RT = R >> 5, CT = C >> 5;
  int blk = blockIdx.x;
  int b = blk / (RT * CT);
  int rem = blk % (RT * CT);
  int r0 = (rem / CT) * 32, c0 = (rem % CT) * 32;
  const float* S = src + (size_t)b * R * C;
  bf16* D = dst + (size_t)b * R * C;
  int tx = threadIdx.x & 31, ty = threadIdx.x >> 5;
#pragma unroll
  for (int i = 0; i < 32; i += 8)
    tile[ty + i][tx] = S[(size_t)(r0 + ty + i) * C + c0 + tx];
  __syncthreads();
#pragma unroll
  for (int i = 0; i < 32; i += 8)
    D[(size_t)(c0 + ty + i) * R + r0 + tx] = f2bf(tile[tx][ty + i]);
}

// ---------- AMLP projections (2x2 tiles): sum-of-squares partials ----------
__global__ __launch_bounds__(128) void k_proj(
    const bf16* __restrict__ xt, const bf16* __restrict__ xa,
    const bf16* __restrict__ xv, const bf16* __restrict__ Wib,
    const bf16* __restrict__ Wqb, const bf16* __restrict__ Wvb,
    const float* __restrict__ bi, const float* __restrict__ bq,
    const float* __restrict__ bv, float* __restrict__ part) {
  int wave = (int)((blockIdx.x * (long)blockDim.x + threadIdx.x) >> 5);
  const int N2 = F2_ / 32;               // 8
  const int perj = (B_ * S_ / 32) * N2;  // 16384
  int j = wave / perj, t = wave % perj;
  int row0 = (t / N2) * 32, col0 = (t % N2) * 32;
  const bf16* X = (j == 0) ? xt : ((j == 1) ? xa : xv);
  const bf16* W = (j == 0) ? Wib : ((j == 1) ? Wqb : Wvb);
  const float* bb = (j == 0) ? bi : ((j == 1) ? bq : bv);
  v8f acc[4] = {};
  gemm_2x2(X, F_, row0, W, F_, col0, F_, acc);
  int lane = threadIdx.x & 31, n = lane & 15;
  float bias0 = bb[col0 + n], bias1 = bb[col0 + 16 + n];
  float ss = 0.f;
#pragma unroll
  for (int r = 0; r < 8; ++r) {
    float f;
    f = acc[0][r] + bias0; ss += f * f;
    f = acc[1][r] + bias1; ss += f * f;
    f = acc[2][r] + bias0; ss += f * f;
    f = acc[3][r] + bias1; ss += f * f;
  }
#pragma unroll
  for (int o = 16; o > 0; o >>= 1) ss += __shfl_xor(ss, o);
  if (lane == 0) part[wave] = ss;
}

// deterministic tree reduction of the 3 partial arrays -> sumsq[0..2]
__global__ __launch_bounds__(256) void k_reduce3(const float* __restrict__ part,
                                                 int n, float* __restrict__ out) {
  __shared__ float sh[256];
  const float* p = part + (size_t)blockIdx.x * n;
  float s = 0.f;
  for (int i = threadIdx.x; i < n; i += 256) s += p[i];
  sh[threadIdx.x] = s;
  __syncthreads();
  for (int st = 128; st > 0; st >>= 1) {
    if ((int)threadIdx.x < st) sh[threadIdx.x] += sh[threadIdx.x + st];
    __syncthreads();
  }
  if (threadIdx.x == 0) out[blockIdx.x] = sh[0];
}

// ---------- recompute f1/f2 tiles, combine norms, pair-sum -> zraw ----------
__global__ __launch_bounds__(128) void k_gz(
    const bf16* __restrict__ xt, const bf16* __restrict__ xa,
    const bf16* __restrict__ Wib, const bf16* __restrict__ Wqb,
    const float* __restrict__ bi, const float* __restrict__ bq,
    const float* __restrict__ sumsq, float* __restrict__ zraw) {
  int wave = (int)((blockIdx.x * (long)blockDim.x + threadIdx.x) >> 5);
  const int Nt = F2_ / 16;
  int row0 = (wave / Nt) * 16, col0 = (wave % Nt) * 16;
  v8f a0 = {}, a1 = {};
  a0 = gemm_1x1(xt, F_, row0, Wib, F_, col0, F_, a0);
  a1 = gemm_1x1(xa, F_, row0, Wqb, F_, col0, F_, a1);
  float n1 = sqrtf(sumsq[0]), n2 = sqrtf(sumsq[1]), n3 = sqrtf(sumsq[2]);
  float c1 = n1 / (n1 + n2), c2 = n2 / (n1 + n2);
  float c3s = n3 * n3 / (n1 + n2 + n3);
  int lane = threadIdx.x & 31, n = lane & 15, hf = lane >> 4;
  float bv1 = bi[col0 + n], bv2 = bq[col0 + n];
#pragma unroll
  for (int r = 0; r < 8; ++r) {
    float p = c1 * (a0[r] + bv1) + c2 * (a1[r] + bv2);
    float q = p + __shfl_xor(p, 1);  // pair-sum over adjacent columns
    if ((n & 1) == 0) {
      int row = row0 + r + hf * 8;
      zraw[(size_t)row * F_ + (col0 >> 1) + (n >> 1)] = q + 2.f * c3s;
    }
  }
}

// deterministic per-(b,f) sum over S of z^2 (coalesced across threads)
__global__ __launch_bounds__(256) void k_znsq(const float* __restrict__ zraw,
                                              float* __restrict__ znsq) {
  int i = blockIdx.x * blockDim.x + threadIdx.x;
  if (i >= B_ * F_) return;
  int b = i / F_, f = i % F_;
  const float* p = zraw + (size_t)b * S_ * F_ + f;
  float s = 0.f;
  for (int sidx = 0; sidx < S_; ++sidx) {
    float v = p[sidx * F_];
    s += v * v;
  }
  znsq[i] = s;
}

// G^T[b,f,s] = z[b,s,f]/max(||z[b,:,f]||,1e-12) ; fused LDS transpose
__global__ __launch_bounds__(256) void k_gnorm_t(const float* __restrict__ zraw,
                                                 const float* __restrict__ znsq,
                                                 bf16* __restrict__ GT) {
  __shared__ float tile[32][33];
  const int ST = S_ >> 5, FT = F_ >> 5;  // 32, 4
  int blk = blockIdx.x;
  int b = blk / (ST * FT);
  int rem = blk % (ST * FT);
  int s0 = (rem / FT) * 32, f0 = (rem % FT) * 32;
  int tx = threadIdx.x & 31, ty = threadIdx.x >> 5;
#pragma unroll
  for (int i = 0; i < 32; i += 8)
    tile[ty + i][tx] = zraw[((size_t)b * S_ + s0 + ty + i) * F_ + f0 + tx];
  __syncthreads();
#pragma unroll
  for (int i = 0; i < 32; i += 8) {
    int f = f0 + ty + i;
    float d = fmaxf(sqrtf(znsq[b * F_ + f]), 1e-12f);
    GT[((size_t)b * F_ + f) * S_ + s0 + tx] = f2bf(tile[tx][ty + i] / d);
  }
}

// ---------- T_m = x_m^T @ A_m  ([F,S] per batch, bf16), 2x2 tiles ----------
__global__ __launch_bounds__(128) void k_T(
    const bf16* __restrict__ xTa, const bf16* __restrict__ xTv,
    const bf16* __restrict__ xTt, const bf16* __restrict__ Ata,
    const bf16* __restrict__ Atv, const bf16* __restrict__ Atl,
    bf16* __restrict__ Ta, bf16* __restrict__ Tv, bf16* __restrict__ Tt) {
  int wave = (int)((blockIdx.x * (long)blockDim.x + threadIdx.x) >> 5);
  const int M2 = F_ / 32, N2 = S_ / 32;  // 4, 32
  const int perB = M2 * N2;              // 128
  const int perM = B_ * perB;            // 8192
  int m = wave / perM, rem = wave % perM;
  int b = rem / perB, t = rem % perB;
  int row0 = (t / N2) * 32, col0 = (t % N2) * 32;
  const bf16* X = (m == 0) ? xTa : ((m == 1) ? xTv : xTt);
  const bf16* At = (m == 0) ? Ata : ((m == 1) ? Atv : Atl);
  bf16* T = (m == 0) ? Ta : ((m == 1) ? Tv : Tt);
  X += (size_t)b * F_ * S_;
  T += (size_t)b * F_ * S_;
  v8f acc[4] = {};
  gemm_2x2(X, S_, row0, At, S_, col0, S_, acc);
  store_tile_bf16(T, S_, row0, col0, acc[0]);
  store_tile_bf16(T, S_, row0, col0 + 16, acc[1]);
  store_tile_bf16(T, S_, row0 + 16, col0, acc[2]);
  store_tile_bf16(T, S_, row0 + 16, col0 + 16, acc[3]);
}

// ---------- C_m = tanh( T_m @ [y_m | G] / 16 )  ([F,2F], bf16), 2x2 ----------
// y: m0->aud, m1->aud (source bug: G_ag reused), m2->txt. col0 mult of 32 so
// both sub-tiles stay in one concat half (F_ = 128 is a multiple of 32).
__global__ __launch_bounds__(128) void k_C(
    const bf16* __restrict__ Ta, const bf16* __restrict__ Tv,
    const bf16* __restrict__ Tt, const bf16* __restrict__ xTa,
    const bf16* __restrict__ xTt, const bf16* __restrict__ GT,
    bf16* __restrict__ Ca, bf16* __restrict__ Cv, bf16* __restrict__ Ct) {
  int wave = (int)((blockIdx.x * (long)blockDim.x + threadIdx.x) >> 5);
  const int M2 = F_ / 32, N2 = F2_ / 32;  // 4, 8
  const int perB = M2 * N2;               // 32
  const int perM = B_ * perB;             // 2048
  int m = wave / perM, rem = wave % perM;
  int b = rem / perB, t = rem % perB;
  int row0 = (t / N2) * 32, col0 = (t % N2) * 32;
  const bf16* T = (m == 0) ? Ta : ((m == 1) ? Tv : Tt);
  bf16* C = (m == 0) ? Ca : ((m == 1) ? Cv : Ct);
  const bf16* YT = (m == 2) ? xTt : xTa;
  T += (size_t)b * F_ * S_;
  C += (size_t)b * F_ * F2_;
  const bf16* Bsrc;
  int bc;
  if (col0 < F_) { Bsrc = YT + (size_t)b * F_ * S_; bc = col0; }
  else           { Bsrc = GT + (size_t)b * F_ * S_; bc = col0 - F_; }
  v8f acc[4] = {};
  gemm_2x2(T, S_, row0, Bsrc, S_, bc, S_, acc);
  int lane = threadIdx.x & 31, n = lane & 15, hf = lane >> 4;
#pragma unroll
  for (int q = 0; q < 4; ++q) {
    int rr = row0 + ((q >> 1) ? 16 : 0);
    int cc = col0 + ((q & 1) ? 16 : 0);
#pragma unroll
    for (int r = 0; r < 8; ++r)
      C[(size_t)(rr + r + hf * 8) * F2_ + cc + n] =
          f2bf(tanhf(acc[q][r] * 0.0625f));  // 1/sqrt(2F) = 1/16
  }
}

// ---------- H_m = relu( C_m Wc^T + x_m^T W^T )  ([F,K] row-major, bf16) ------
__global__ __launch_bounds__(128) void k_H(
    const bf16* __restrict__ Ca, const bf16* __restrict__ Cv,
    const bf16* __restrict__ Ct, const bf16* __restrict__ Wcba,
    const bf16* __restrict__ Wcbv, const bf16* __restrict__ Wcbt,
    const bf16* __restrict__ xTa, const bf16* __restrict__ xTv,
    const bf16* __restrict__ xTt, const bf16* __restrict__ Wxba,
    const bf16* __restrict__ Wxbv, const bf16* __restrict__ Wxbt,
    bf16* __restrict__ Ha, bf16* __restrict__ Hv, bf16* __restrict__ Ht) {
  int wave = (int)((blockIdx.x * (long)blockDim.x + threadIdx.x) >> 5);
  const int Mt = F_ / 16, Nt = K_ / 16;  // 8, 4
  const int perB = Mt * Nt;              // 32
  const int perM = B_ * perB;            // 2048
  int m = wave / perM, rem = wave % perM;
  int b = rem / perB, t = rem % perB;
  int row0 = (t / Nt) * 16, col0 = (t % Nt) * 16;
  const bf16* C = (m == 0) ? Ca : ((m == 1) ? Cv : Ct);
  const bf16* Wcb = (m == 0) ? Wcba : ((m == 1) ? Wcbv : Wcbt);
  const bf16* XT = (m == 0) ? xTa : ((m == 1) ? xTv : xTt);
  const bf16* Wxb = (m == 0) ? Wxba : ((m == 1) ? Wxbv : Wxbt);
  bf16* H = (m == 0) ? Ha : ((m == 1) ? Hv : Ht);
  C += (size_t)b * F_ * F2_;
  XT += (size_t)b * F_ * S_;
  H += (size_t)b * F_ * K_;
  v8f acc = {};
  acc = gemm_1x1(C, F2_, row0, Wcb, F2_, col0, F2_, acc);
  acc = gemm_1x1(XT, S_, row0, Wxb, S_, col0, S_, acc);
  int lane = threadIdx.x & 31, n = lane & 15, hf = lane >> 4;
#pragma unroll
  for (int r = 0; r < 8; ++r)
    H[(size_t)(row0 + r + hf * 8) * K_ + col0 + n] = f2bf(fmaxf(acc[r], 0.f));
}

// ---------- out_m = Wh_m @ H_m^T + x_m  (f32 epilogue to d_out), 2x2 ---------
// Slots: m0->1 (a), m1->2 (v), m2->0 (t).
__global__ __launch_bounds__(128) void k_out(
    const bf16* __restrict__ Whba, const bf16* __restrict__ Whbv,
    const bf16* __restrict__ Whbt, const bf16* __restrict__ Ha,
    const bf16* __restrict__ Hv, const bf16* __restrict__ Ht,
    const float* __restrict__ auds, const float* __restrict__ viss,
    const float* __restrict__ txts, float* __restrict__ out) {
  int wave = (int)((blockIdx.x * (long)blockDim.x + threadIdx.x) >> 5);
  const int M2 = S_ / 32, N2 = F_ / 32;  // 32, 4
  const int perB = M2 * N2;              // 128
  const int perM = B_ * perB;            // 8192
  int m = wave / perM, rem = wave % perM;
  int b = rem / perB, t = rem % perB;
  int row0 = (t / N2) * 32, col0 = (t % N2) * 32;
  const bf16* Wh = (m == 0) ? Whba : ((m == 1) ? Whbv : Whbt);
  const bf16* H = (m == 0) ? Ha : ((m == 1) ? Hv : Ht);
  const float* res = (m == 0) ? auds : ((m == 1) ? viss : txts);
  int slot = (m == 0) ? 1 : ((m == 1) ? 2 : 0);
  H += (size_t)b * F_ * K_;
  v8f acc[4] = {};
  gemm_2x2(Wh, K_, row0, H, K_, col0, K_, acc);
  int lane = threadIdx.x & 31, n = lane & 15, hf = lane >> 4;
  size_t obase = (size_t)slot * B_ * S_ * F_;
#pragma unroll
  for (int q = 0; q < 4; ++q) {
    int rr = row0 + ((q >> 1) ? 16 : 0);
    int cc = col0 + ((q & 1) ? 16 : 0);
#pragma unroll
    for (int r = 0; r < 8; ++r) {
      int srow = rr + r + hf * 8;
      size_t idx = ((size_t)b * S_ + srow) * F_ + cc + n;
      out[obase + idx] = acc[q][r] + res[idx];
    }
  }
}

// ---------------------------------------------------------------------------
extern "C" void kernel_launch(void* const* d_in, const int* in_sizes, int n_in,
                              void* d_out, int out_size, void* d_ws,
                              size_t ws_size, hipStream_t stream) {
  (void)in_sizes; (void)n_in; (void)out_size; (void)ws_size;
  const float* txt = (const float*)d_in[0];
  const float* aud = (const float*)d_in[1];
  const float* vis = (const float*)d_in[2];
  const float* Wi  = (const float*)d_in[3];
  const float* bi  = (const float*)d_in[4];
  const float* Wq  = (const float*)d_in[5];
  const float* bq  = (const float*)d_in[6];
  const float* Wvp = (const float*)d_in[7];
  const float* bvp = (const float*)d_in[8];
  const float* Aa  = (const float*)d_in[9];
  const float* Av  = (const float*)d_in[10];
  const float* Al  = (const float*)d_in[11];
  const float* Wa  = (const float*)d_in[12];
  const float* Wv  = (const float*)d_in[13];
  const float* Wt  = (const float*)d_in[14];
  const float* Wca = (const float*)d_in[15];
  const float* Wcv = (const float*)d_in[16];
  const float* Wct = (const float*)d_in[17];
  const float* Wha = (const float*)d_in[18];
  const float* Whv = (const float*)d_in[19];
  const float* Wht = (const float*)d_in[20];
  float* out = (float*)d_out;

  const size_t BSF = (size_t)B_ * S_ * F_;
  uint8_t* base = (uint8_t*)d_ws;
  size_t off = 0;
  auto alloc = [&](size_t bytes) -> void* {
    off = (off + 255) & ~(size_t)255;
    void* p = base + off;
    off += bytes;
    return p;
  };
  float* scal = (float*)alloc(16 * sizeof(float));
  float* part = (float*)alloc((size_t)3 * 16384 * sizeof(float));
  float* znsq = (float*)alloc((size_t)B_ * F_ * sizeof(float));
  float* zraw = (float*)alloc(BSF * sizeof(float));
  bf16* xbt = (bf16*)alloc(BSF * 2);   // x row-major bf16 (proj A-op)
  bf16* xba = (bf16*)alloc(BSF * 2);
  bf16* xbv = (bf16*)alloc(BSF * 2);
  bf16* xTa = (bf16*)alloc(BSF * 2);   // x^T [B,F,S]
  bf16* xTv = (bf16*)alloc(BSF * 2);
  bf16* xTt = (bf16*)alloc(BSF * 2);
  bf16* Wib = (bf16*)alloc((size_t)F2_ * F_ * 2);  // original [2F,F] layouts
  bf16* Wqb = (bf16*)alloc((size_t)F2_ * F_ * 2);
  bf16* Wvb = (bf16*)alloc((size_t)F2_ * F_ * 2);
  bf16* Ata = (bf16*)alloc((size_t)S_ * S_ * 2);   // A^T [s,t]
  bf16* Atv = (bf16*)alloc((size_t)S_ * S_ * 2);
  bf16* Atl = (bf16*)alloc((size_t)S_ * S_ * 2);
  bf16* Wcba = (bf16*)alloc((size_t)K_ * F2_ * 2); // original [K,2F]
  bf16* Wcbv = (bf16*)alloc((size_t)K_ * F2_ * 2);
  bf16* Wcbt = (bf16*)alloc((size_t)K_ * F2_ * 2);
  bf16* Wxba = (bf16*)alloc((size_t)K_ * S_ * 2);  // original [K,S]
  bf16* Wxbv = (bf16*)alloc((size_t)K_ * S_ * 2);
  bf16* Wxbt = (bf16*)alloc((size_t)K_ * S_ * 2);
  bf16* Whba = (bf16*)alloc((size_t)S_ * K_ * 2);  // original [S,K]
  bf16* Whbv = (bf16*)alloc((size_t)S_ * K_ * 2);
  bf16* Whbt = (bf16*)alloc((size_t)S_ * K_ * 2);
  bf16* GT  = (bf16*)alloc(BSF * 2);   // G^T [B,F,S]
  bf16* Ta  = (bf16*)alloc(BSF * 2);   // T [B,F,S]
  bf16* Tv  = (bf16*)alloc(BSF * 2);
  bf16* Tt  = (bf16*)alloc(BSF * 2);
  bf16* Cam = (bf16*)alloc((size_t)B_ * F_ * F2_ * 2);
  bf16* Cvm = (bf16*)alloc((size_t)B_ * F_ * F2_ * 2);
  bf16* Ctm = (bf16*)alloc((size_t)B_ * F_ * F2_ * 2);
  bf16* Ha  = (bf16*)alloc((size_t)B_ * F_ * K_ * 2);
  bf16* Hv  = (bf16*)alloc((size_t)B_ * F_ * K_ * 2);
  bf16* Htm = (bf16*)alloc((size_t)B_ * F_ * K_ * 2);

  // --- plain conversions (f32 -> bf16, original layouts, vectorized) ---
  k_cvt_plain<<<2048, 256, 0, stream>>>(txt, xbt, (long)(BSF / 4));
  k_cvt_plain<<<2048, 256, 0, stream>>>(aud, xba, (long)(BSF / 4));
  k_cvt_plain<<<2048, 256, 0, stream>>>(vis, xbv, (long)(BSF / 4));
  k_cvt_plain<<<32, 256, 0, stream>>>(Wi, Wib, (long)(F2_ * F_ / 4));
  k_cvt_plain<<<32, 256, 0, stream>>>(Wq, Wqb, (long)(F2_ * F_ / 4));
  k_cvt_plain<<<32, 256, 0, stream>>>(Wvp, Wvb, (long)(F2_ * F_ / 4));
  k_cvt_plain<<<16, 256, 0, stream>>>(Wca, Wcba, (long)(K_ * F2_ / 4));
  k_cvt_plain<<<16, 256, 0, stream>>>(Wcv, Wcbv, (long)(K_ * F2_ / 4));
  k_cvt_plain<<<16, 256, 0, stream>>>(Wct, Wcbt, (long)(K_ * F2_ / 4));
  k_cvt_plain<<<64, 256, 0, stream>>>(Wa, Wxba, (long)(K_ * S_ / 4));
  k_cvt_plain<<<64, 256, 0, stream>>>(Wv, Wxbv, (long)(K_ * S_ / 4));
  k_cvt_plain<<<64, 256, 0, stream>>>(Wt, Wxbt, (long)(K_ * S_ / 4));
  k_cvt_plain<<<64, 256, 0, stream>>>(Wha, Whba, (long)(S_ * K_ / 4));
  k_cvt_plain<<<64, 256, 0, stream>>>(Whv, Whbv, (long)(S_ * K_ / 4));
  k_cvt_plain<<<64, 256, 0, stream>>>(Wht, Whbt, (long)(S_ * K_ / 4));

  // --- LDS-tiled transposes ---
  k_tr_cvt<<<8192, 256, 0, stream>>>(aud, xTa, S_, F_);  // [B,S,F]->[B,F,S]
  k_tr_cvt<<<8192, 256, 0, stream>>>(vis, xTv, S_, F_);
  k_tr_cvt<<<8192, 256, 0, stream>>>(txt, xTt, S_, F_);
  k_tr_cvt<<<1024, 256, 0, stream>>>(Aa, Ata, S_, S_);   // [t,s]->[s,t]
  k_tr_cvt<<<1024, 256, 0, stream>>>(Av, Atv, S_, S_);
  k_tr_cvt<<<1024, 256, 0, stream>>>(Al, Atl, S_, S_);

  // --- AMLP: projections + global norms (deterministic reduction) ---
  k_proj<<<12288, 128, 0, stream>>>(xbt, xba, xbv, Wib, Wqb, Wvb, bi, bq, bvp,
                                    part);
  k_reduce3<<<3, 256, 0, stream>>>(part, 16384, scal);
  k_gz<<<16384, 128, 0, stream>>>(xbt, xba, Wib, Wqb, bi, bq, scal, zraw);
  k_znsq<<<(B_ * F_ + 255) / 256, 256, 0, stream>>>(zraw, znsq);
  k_gnorm_t<<<8192, 256, 0, stream>>>(zraw, znsq, GT);

  // --- fusion trunk ---
  k_T<<<6144, 128, 0, stream>>>(xTa, xTv, xTt, Ata, Atv, Atl, Ta, Tv, Tt);
  k_C<<<1536, 128, 0, stream>>>(Ta, Tv, Tt, xTa, xTt, GT, Cam, Cvm, Ctm);
  k_H<<<1536, 128, 0, stream>>>(Cam, Cvm, Ctm, Wcba, Wcbv, Wcbt, xTa, xTv, xTt,
                                Wxba, Wxbv, Wxbt, Ha, Hv, Htm);
  k_out<<<6144, 128, 0, stream>>>(Whba, Whbv, Whbt, Ha, Hv, Htm, aud, vis,
                                  txt, out);
}
